// PointNetPP_24824910971383
// MI455X (gfx1250) — matrix operand order repaired and checked by reading, not compiled, not run on hardware
//
#include <hip/hip_runtime.h>
#include <hip/hip_bf16.h>
#include <math.h>

typedef _Float16 v16h __attribute__((ext_vector_type(16)));
typedef _Float16 v8h  __attribute__((ext_vector_type(8)));
typedef float    v8f  __attribute__((ext_vector_type(8)));

#define BG   4
#define N0   8192
#define M1   4096
#define M2   1024
#define EPQ  33   // 32 radius candidates + 1 appended self loop

// ---------------------------------------------------------------- utilities

__device__ inline void atomicMaxF32(float* addr, float val) {
  unsigned int* ua = (unsigned int*)addr;
  unsigned int old = *ua;
  while (true) {
    float f = __uint_as_float(old);
    if (f >= val) break;
    unsigned int assumed = old;
    old = atomicCAS(ua, assumed, __float_as_uint(val));
    if (old == assumed) break;
  }
}

__global__ void k_fill_f32(float* __restrict__ p, long long n, float v) {
  long long t = (long long)blockIdx.x * blockDim.x + threadIdx.x;
  if (t < n) p[t] = v;
}

__global__ void k_fill_f16(_Float16* __restrict__ p, long long n) {
  long long t = (long long)blockIdx.x * blockDim.x + threadIdx.x;
  if (t < n) p[t] = (_Float16)0.0f;
}

// Pack weights [K x N] f32 -> WMMA B-fragment order f16:
//   dst[((c*nT + t)*32 + lane)*16 + j]  where k = c*32 + (lane>>4)*16 + j, n = t*16 + (lane&15)
// Each lane's 16 halves for one (chunk, tile) become one contiguous 32B load.
__global__ void k_weight_pack(_Float16* __restrict__ dst, const float* __restrict__ src,
                              int K, int N, int Kp, int Np) {
  int nT = Np >> 4;
  long long tot = (long long)Kp * Np;
  long long t = (long long)blockIdx.x * blockDim.x + threadIdx.x;
  if (t >= tot) return;
  int j = (int)(t & 15);
  int lane = (int)((t >> 4) & 31);
  long long rest = t >> 9;
  int tIdx = (int)(rest % nT);
  int c = (int)(rest / nT);
  int k = c * 32 + ((lane >> 4) << 4) + j;
  int n = tIdx * 16 + (lane & 15);
  float v = (k < K && n < N) ? src[(size_t)k * N + n] : 0.0f;
  dst[t] = (_Float16)v;
}

__global__ void k_bias_cvt(float* __restrict__ dst, const float* __restrict__ src,
                           int N, int Np) {
  int t = blockIdx.x * blockDim.x + threadIdx.x;
  if (t >= Np) return;
  dst[t] = (t < N) ? src[t] : 0.0f;
}

// x0 = concat(x[6], pos[2]) -> f32 [B*N0, 8]
__global__ void k_concat_x0(float* __restrict__ x0f, const float* __restrict__ x,
                            const float* __restrict__ pos, int total) {
  int r = blockIdx.x * blockDim.x + threadIdx.x;
  if (r >= total) return;
  #pragma unroll
  for (int c = 0; c < 6; ++c) x0f[(size_t)r * 8 + c] = x[(size_t)r * 6 + c];
  x0f[(size_t)r * 8 + 6] = pos[(size_t)r * 2 + 0];
  x0f[(size_t)r * 8 + 7] = pos[(size_t)r * 2 + 1];
}

// ---------------------------------------------------------------- FPS (one block per graph, LDS-staged)

__global__ void k_fps(int* __restrict__ idxOut, const float* __restrict__ pos, int n, int m) {
  extern __shared__ float sm[];
  float* sx = sm;
  float* sy = sm + n;
  float* sd = sm + 2 * n;
  __shared__ float sval[512];
  __shared__ int   sidx[512];
  __shared__ float q[2];
  int g = blockIdx.x, tid = threadIdx.x;
  const float* pg = pos + (size_t)g * n * 2;
  for (int p = tid; p < n; p += blockDim.x) {
    sx[p] = pg[p * 2 + 0];
    sy[p] = pg[p * 2 + 1];
    sd[p] = 3.0e38f;
  }
  if (tid == 0) { q[0] = pg[0]; q[1] = pg[1]; }
  __syncthreads();
  int last = 0;
  for (int j = 0; j < m; ++j) {
    if (tid == 0) idxOut[(size_t)g * m + j] = last;
    float qx = q[0], qy = q[1];
    float bv = -1.0f; int bi = 0x7fffffff;
    for (int p = tid; p < n; p += blockDim.x) {
      float dx = sx[p] - qx, dy = sy[p] - qy;
      float d = dx * dx + dy * dy;
      float dm = sd[p];
      dm = d < dm ? d : dm;
      sd[p] = dm;
      if (dm > bv || (dm == bv && p < bi)) { bv = dm; bi = p; }
    }
    sval[tid] = bv; sidx[tid] = bi;
    __syncthreads();
    for (int s = 256; s > 0; s >>= 1) {
      if (tid < s) {
        float ov = sval[tid + s]; int oi = sidx[tid + s];
        if (ov > sval[tid] || (ov == sval[tid] && oi < sidx[tid])) {
          sval[tid] = ov; sidx[tid] = oi;
        }
      }
      __syncthreads();
    }
    last = sidx[0];
    if (tid == 0) { q[0] = sx[last]; q[1] = sy[last]; }
    __syncthreads();
  }
}

__global__ void k_gather_pos(float* __restrict__ dst, const float* __restrict__ src,
                             const int* __restrict__ idx, int m, int n, int Bg) {
  int t = blockIdx.x * blockDim.x + threadIdx.x;
  if (t >= Bg * m) return;
  int g = t / m;
  int pi = idx[t];
  dst[(size_t)t * 2 + 0] = src[((size_t)g * n + pi) * 2 + 0];
  dst[(size_t)t * 2 + 1] = src[((size_t)g * n + pi) * 2 + 1];
}

// ---------------------------------------------------------------- radius / top-K (K=32) per query

__global__ void k_topk(int* __restrict__ cols, unsigned char* __restrict__ valid,
                       const float* __restrict__ pos, const int* __restrict__ idx,
                       int Bg, int n, int m, float r2) {
  int t = blockIdx.x * blockDim.x + threadIdx.x;
  if (t >= Bg * m) return;
  int g = t / m, q = t - g * m;
  const float* pg = pos + (size_t)g * n * 2;
  int pi = idx[t];
  float qx = pg[pi * 2 + 0], qy = pg[pi * 2 + 1];
  float kd[32]; int ki[32];
  #pragma unroll
  for (int i = 0; i < 32; ++i) { kd[i] = 3.0e38f; ki[i] = 0; }
  for (int j = 0; j < n; ++j) {
    float dx = pg[j * 2 + 0] - qx, dy = pg[j * 2 + 1] - qy;
    float d = dx * dx + dy * dy;
    if (d < kd[31]) {
      int p = 31;
      while (p > 0 && kd[p - 1] > d) { kd[p] = kd[p - 1]; ki[p] = ki[p - 1]; --p; }
      kd[p] = d; ki[p] = j;
    }
  }
  size_t base = (size_t)t * EPQ;
  for (int s = 0; s < 32; ++s) {
    cols[base + s] = ki[s];
    valid[base + s] = (kd[s] <= r2 && ki[s] != q) ? 1 : 0;   // PyG quirk: col vs row-index
  }
  cols[base + 32] = q;      // appended self loop (i,i) with row-index semantics
  valid[base + 32] = 1;
}

// edge feature gather: concat(x[col], pos[col]-pos[query]) -> f16 matrix row
__global__ void k_gather_sa(_Float16* __restrict__ ein, int lda,
                            const float* __restrict__ xsrc, int Cin,
                            const float* __restrict__ psrc,
                            const int* __restrict__ qidx,
                            const int* __restrict__ cols, int m) {
  int C2 = Cin + 2;
  long long tot = (long long)m * EPQ * C2;
  long long t = (long long)blockIdx.x * blockDim.x + threadIdx.x;
  if (t >= tot) return;
  int c = (int)(t % C2);
  long long e = t / C2;
  int q = (int)(e / EPQ);
  int col = cols[e];
  float v;
  if (c < Cin) {
    v = xsrc[(size_t)col * Cin + c];
  } else {
    int pi = qidx[q];
    int d = c - Cin;
    v = psrc[(size_t)col * 2 + d] - psrc[(size_t)pi * 2 + d];
  }
  ein[(size_t)e * lda + c] = (_Float16)v;
}

// ---------------------------------------------------------------- kNN interpolate (search + weighted gather)

__global__ void k_knn_search(int* __restrict__ kidx, float* __restrict__ kw,
                             const float* __restrict__ ptgt, const float* __restrict__ psrc,
                             int Bg, int mt, int ms, int k) {
  int t = blockIdx.x * blockDim.x + threadIdx.x;
  if (t >= Bg * mt) return;
  int g = t / mt;
  float qx = ptgt[(size_t)t * 2 + 0], qy = ptgt[(size_t)t * 2 + 1];
  const float* ps = psrc + (size_t)g * ms * 2;
  float kd[16]; int ki[16];
  #pragma unroll
  for (int i = 0; i < 16; ++i) { kd[i] = 3.0e38f; ki[i] = 0; }
  for (int j = 0; j < ms; ++j) {
    float dx = ps[j * 2 + 0] - qx, dy = ps[j * 2 + 1] - qy;
    float d = dx * dx + dy * dy;
    if (d < kd[k - 1]) {
      int p = k - 1;
      while (p > 0 && kd[p - 1] > d) { kd[p] = kd[p - 1]; ki[p] = ki[p - 1]; --p; }
      kd[p] = d; ki[p] = j;
    }
  }
  float sw = 0.0f;
  for (int j = 0; j < k; ++j) sw += 1.0f / fmaxf(kd[j], 1e-16f);
  for (int j = 0; j < k; ++j) {
    kidx[(size_t)t * k + j] = g * ms + ki[j];
    kw[(size_t)t * k + j] = (1.0f / fmaxf(kd[j], 1e-16f)) / sw;
  }
}

__global__ void k_u_gather(_Float16* __restrict__ dst, int ldd, int colOff,
                           const float* __restrict__ src, int lds, int C,
                           const int* __restrict__ kidx, const float* __restrict__ kw,
                           int k, int T) {
  long long t = (long long)blockIdx.x * blockDim.x + threadIdx.x;
  if (t >= (long long)T * C) return;
  int c = (int)(t % C);
  int r = (int)(t / C);
  float acc = 0.0f;
  for (int j = 0; j < k; ++j)
    acc += kw[(size_t)r * k + j] * src[(size_t)kidx[(size_t)r * k + j] * lds + c];
  dst[(size_t)r * ldd + colOff + c] = (_Float16)acc;
}

__global__ void k_cvt_cols(_Float16* __restrict__ dst, int ldd, int colOff,
                           const float* __restrict__ src, int lds, int rows, int C) {
  long long t = (long long)blockIdx.x * blockDim.x + threadIdx.x;
  if (t >= (long long)rows * C) return;
  int c = (int)(t % C);
  int r = (int)(t / C);
  dst[(size_t)r * ldd + colOff + c] = (_Float16)src[(size_t)r * lds + c];
}

// u3 broadcast: each row takes its graph's global-max vector
__global__ void k_bcast_x3(_Float16* __restrict__ dst, int ldd, const float* __restrict__ x3,
                           int rows, int C, int perG) {
  long long t = (long long)blockIdx.x * blockDim.x + threadIdx.x;
  if (t >= (long long)rows * C) return;
  int c = (int)(t % C);
  int r = (int)(t / C);
  dst[(size_t)r * ldd + c] = (_Float16)x3[(size_t)(r / perG) * C + c];
}

// ---------------------------------------------------------------- WMMA GEMM
// C = A(f16)[MxK, lda padded] * B(packed f16 fragments) + bias
// Each wave computes a 16 x (16*NT) slab. Per K-chunk: one A fragment (2x b128),
// NT B fragments loaded up-front in one clause (immediate offsets off a running
// pointer), then NT back-to-back WMMAs. B pointer strength-reduced: one uniform
// 64-bit add per chunk, no per-tile address rebuild.
// mode 0: tanh -> f16   mode 1: f32 store (col<Nreal)   mode 2: atomic max-aggregate per row group

template <int NT>
__global__ void k_wmma_gemm(const _Float16* __restrict__ A, int lda,
                            const _Float16* __restrict__ Bp, int bnT,  // bnT = Np/16
                            const float* __restrict__ bias,
                            int M, int N, int K, int Nreal, int mode,
                            _Float16* __restrict__ outH, float* __restrict__ outF, int ldo,
                            const unsigned char* __restrict__ valid, int gdiv,
                            float* __restrict__ aggr, int ald) {
  int wave = (int)(((unsigned)blockIdx.x * blockDim.x + threadIdx.x) >> 5);
  int lane = threadIdx.x & 31;
  int tilesM = M >> 4;
  int groupsN = (N >> 4) / NT;
  if (wave >= tilesM * groupsN) return;    // wave-uniform: EXEC stays all-ones for WMMA
  int tm = wave % tilesM;
  int tn0 = (wave / tilesM) * NT;
  int hsel = lane >> 4;
  int l15 = lane & 15;
  v8f acc[NT];
  #pragma unroll
  for (int t = 0; t < NT; ++t) acc[t] = (v8f){0.f, 0.f, 0.f, 0.f, 0.f, 0.f, 0.f, 0.f};
  // A fragment (16-bit 16x32): lanes 0-15 hold K 0-15; lanes 16-31 hold K 16-31
  const _Float16* ap = A + (size_t)(tm * 16 + l15) * lda + (hsel ? 16 : 0);
  // B fragment pointer: tile offsets are compile-time immediates (t*512 halves);
  // advance once per chunk by bstride (uniform).
  const _Float16* bp = Bp + ((size_t)tn0 * 32 + (size_t)lane) * 16;
  const size_t bstride = (size_t)bnT * 512;
  for (int k0 = 0; k0 < K; k0 += 32) {
    __builtin_prefetch(ap + k0 + 64, 0, 1);
    v8h lo = *(const v8h*)(ap + k0);
    v8h hi = *(const v8h*)(ap + k0 + 8);
    v16h af = __builtin_shufflevector(lo, hi, 0, 1, 2, 3, 4, 5, 6, 7,
                                      8, 9, 10, 11, 12, 13, 14, 15);
    v16h bf[NT];
    #pragma unroll
    for (int t = 0; t < NT; ++t) bf[t] = *(const v16h*)(bp + (size_t)t * 512);
    #pragma unroll
    for (int t = 0; t < NT; ++t)
      acc[t] = __builtin_amdgcn_wmma_f32_16x16x32_f16(false, af, false, bf[t],
                                                      (short)0, acc[t], false, false);
    bp += bstride;
  }
  int rbase = tm * 16 + (hsel ? 8 : 0);
  #pragma unroll
  for (int t = 0; t < NT; ++t) {
    int col = (tn0 + t) * 16 + l15;
    float bb = bias ? bias[col] : 0.0f;
    #pragma unroll
    for (int i = 0; i < 8; ++i) {
      int r = rbase + i;
      float v = acc[t][i] + bb;
      if (mode == 0) {
        outH[(size_t)r * ldo + col] = (_Float16)tanhf(v);
      } else if (mode == 1) {
        if (col < Nreal) outF[(size_t)r * ldo + col] = v;
      } else {
        if (col < Nreal && (valid == nullptr || valid[r] != 0)) {
          int qq = r / gdiv;
          atomicMaxF32(&aggr[(size_t)qq * ald + col], v);
        }
      }
    }
  }
}

// ---------------------------------------------------------------- host orchestration

extern "C" void kernel_launch(void* const* d_in, const int* in_sizes, int n_in,
                              void* d_out, int out_size, void* d_ws, size_t ws_size,
                              hipStream_t stream) {
  (void)out_size; (void)ws_size;
  const float* x   = (const float*)d_in[0];
  const float* pos = (const float*)d_in[1];

  // ---- param pointers (disambiguate flattening order via in_sizes) ----
  const float* P[22];
  for (int i = 0; i < 22; ++i) P[i] = (3 + i < n_in) ? (const float*)d_in[3 + i] : nullptr;
  bool ins = (in_sizes[3] == 10 * 64);  // insertion order: mlp1,mlp2,mlp3,fp3,fp2,fp1
  #define GETP(ia, ib) (ins ? P[(ia)] : P[(ib)])
  const float *W1a = GETP(0, 10), *b1a = GETP(1, 11), *W1b = GETP(2, 12), *b1b = GETP(3, 13);
  const float *W2a = GETP(4, 14), *b2a = GETP(5, 15), *W2b = GETP(6, 16), *b2b = GETP(7, 17);
  const float *W3a = GETP(8, 18), *b3a = GETP(9, 19), *W3b = GETP(10, 20), *b3b = GETP(11, 21);
  const float *Wf3 = GETP(12, 8),  *bf3 = GETP(13, 9);
  const float *Wf2 = GETP(14, 6),  *bf2 = GETP(15, 7);
  const float *Wf1a = GETP(16, 0), *bf1a = GETP(17, 1);
  const float *Wf1b = GETP(18, 2), *bf1b = GETP(19, 3);
  const float *Wf1c = GETP(20, 4), *bf1c = GETP(21, 5);

  // ---- workspace carve-up ----
  char* cur = (char*)d_ws;
  auto alloc = [&](size_t bytes) -> void* {
    void* p = (void*)cur;
    cur += (bytes + 255) & ~(size_t)255;
    return p;
  };
  _Float16* W1aH = (_Float16*)alloc(32 * 64 * 2);
  _Float16* W1bH = (_Float16*)alloc(64 * 128 * 2);
  _Float16* W2aH = (_Float16*)alloc(160 * 128 * 2);
  _Float16* W2bH = (_Float16*)alloc(128 * 256 * 2);
  _Float16* W3aH = (_Float16*)alloc(288 * 256 * 2);
  _Float16* W3bH = (_Float16*)alloc(256 * 1024 * 2);
  _Float16* Wf3H = (_Float16*)alloc(1280 * 256 * 2);
  _Float16* Wf2H = (_Float16*)alloc(384 * 128 * 2);
  _Float16* Wf1aH = (_Float16*)alloc(160 * 128 * 2);
  _Float16* Wf1bH = (_Float16*)alloc(128 * 128 * 2);
  _Float16* Wf1cH = (_Float16*)alloc(128 * 16 * 2);
  float* b1aP = (float*)alloc(64 * 4);
  float* b1bP = (float*)alloc(128 * 4);
  float* b2aP = (float*)alloc(128 * 4);
  float* b2bP = (float*)alloc(256 * 4);
  float* b3aP = (float*)alloc(256 * 4);
  float* b3bP = (float*)alloc(1024 * 4);
  float* bf3P = (float*)alloc(256 * 4);
  float* bf2P = (float*)alloc(128 * 4);
  float* bf1aP = (float*)alloc(128 * 4);
  float* bf1bP = (float*)alloc(128 * 4);
  float* bf1cP = (float*)alloc(16 * 4);

  float* x0f = (float*)alloc((size_t)BG * N0 * 8 * 4);
  int*   idx1 = (int*)alloc((size_t)BG * M1 * 4);
  int*   idx2 = (int*)alloc((size_t)BG * M2 * 4);
  float* p1 = (float*)alloc((size_t)BG * M1 * 2 * 4);
  float* p2 = (float*)alloc((size_t)BG * M2 * 2 * 4);
  float* x1f = (float*)alloc((size_t)BG * M1 * 128 * 4);
  float* x2f = (float*)alloc((size_t)BG * M2 * 256 * 4);
  float* x3f = (float*)alloc((size_t)BG * 1024 * 4);
  float* d3f = (float*)alloc((size_t)BG * M2 * 256 * 4);
  float* d2f = (float*)alloc((size_t)BG * M1 * 128 * 4);
  int*   cols1 = (int*)alloc((size_t)BG * M1 * EPQ * 4);
  unsigned char* valid1 = (unsigned char*)alloc((size_t)BG * M1 * EPQ);
  int*   cols2 = (int*)alloc((size_t)BG * M2 * EPQ * 4);
  unsigned char* valid2 = (unsigned char*)alloc((size_t)BG * M2 * EPQ);
  int*   kidx = (int*)alloc((size_t)BG * N0 * 16 * 4);
  float* kw   = (float*)alloc((size_t)BG * N0 * 16 * 4);
  const long long EINB = 6291456;   // halves: max input-matrix footprint (16384x384)
  const long long HABN = 8650752;   // halves: max hidden footprint (135168x64)
  const long long HBBN = 4194304;   // halves: fp1 second hidden (32768x128)
  _Float16* einbuf = (_Float16*)alloc((size_t)EINB * 2);
  _Float16* hA = (_Float16*)alloc((size_t)HABN * 2);
  _Float16* hB = (_Float16*)alloc((size_t)HBBN * 2);

  const float NEGINF = -INFINITY;
  auto blocks = [](long long n) { return (unsigned)((n + 255) / 256); };

  auto gemm = [&](const _Float16* A, int lda, const _Float16* Bp, int Np,
                  const float* bias, int M, int N, int K, int Nreal, int mode,
                  _Float16* oh, float* of, int ldo,
                  const unsigned char* valid, int gdiv, float* aggr, int ald) {
    int bnT = Np >> 4;
    if ((N & 63) == 0) {
      int tiles = (M / 16) * (N / 64);
      int nb = (tiles + 3) / 4;
      k_wmma_gemm<4><<<nb, 128, 0, stream>>>(A, lda, Bp, bnT, bias, M, N, K, Nreal, mode,
                                             oh, of, ldo, valid, gdiv, aggr, ald);
    } else {
      int tiles = (M / 16) * (N / 16);
      int nb = (tiles + 3) / 4;
      k_wmma_gemm<1><<<nb, 128, 0, stream>>>(A, lda, Bp, bnT, bias, M, N, K, Nreal, mode,
                                             oh, of, ldo, valid, gdiv, aggr, ald);
    }
  };
  auto wcvt = [&](const float* W, const float* b, int K, int N, int Kp, int Np,
                  _Float16* Wh, float* bp) {
    k_weight_pack<<<blocks((long long)Kp * Np), 256, 0, stream>>>(Wh, W, K, N, Kp, Np);
    k_bias_cvt<<<blocks(Np), 256, 0, stream>>>(bp, b, N, Np);
  };

  // ---- weights -> packed f16 fragments ----
  wcvt(W1a, b1a, 10, 64, 32, 64, W1aH, b1aP);
  wcvt(W1b, b1b, 64, 128, 64, 128, W1bH, b1bP);
  wcvt(W2a, b2a, 130, 128, 160, 128, W2aH, b2aP);
  wcvt(W2b, b2b, 128, 256, 128, 256, W2bH, b2bP);
  wcvt(W3a, b3a, 258, 256, 288, 256, W3aH, b3aP);
  wcvt(W3b, b3b, 256, 1024, 256, 1024, W3bH, b3bP);
  wcvt(Wf3, bf3, 1280, 256, 1280, 256, Wf3H, bf3P);
  wcvt(Wf2, bf2, 384, 128, 384, 128, Wf2H, bf2P);
  wcvt(Wf1a, bf1a, 136, 128, 160, 128, Wf1aH, bf1aP);
  wcvt(Wf1b, bf1b, 128, 128, 128, 128, Wf1bH, bf1bP);
  wcvt(Wf1c, bf1c, 128, 3, 128, 16, Wf1cH, bf1cP);

  // ---- x0 = concat(x, pos) ----
  k_concat_x0<<<blocks(BG * N0), 256, 0, stream>>>(x0f, x, pos, BG * N0);

  // ---- SA level 1 ----
  k_fps<<<BG, 512, 3 * N0 * 4, stream>>>(idx1, pos, N0, M1);
  k_gather_pos<<<blocks(BG * M1), 256, 0, stream>>>(p1, pos, idx1, M1, N0, BG);
  k_topk<<<blocks(BG * M1), 256, 0, stream>>>(cols1, valid1, pos, idx1, BG, N0, M1, 0.04f);
  k_fill_f16<<<blocks(EINB), 256, 0, stream>>>(einbuf, EINB);
  k_fill_f32<<<blocks((long long)BG * M1 * 128), 256, 0, stream>>>(x1f, (long long)BG * M1 * 128, NEGINF);
  for (int g = 0; g < BG; ++g) {
    int E = M1 * EPQ;   // 135168
    k_gather_sa<<<blocks((long long)E * 10), 256, 0, stream>>>(
        einbuf, 32, x0f + (size_t)g * N0 * 8, 8, pos + (size_t)g * N0 * 2,
        idx1 + (size_t)g * M1, cols1 + (size_t)g * M1 * EPQ, M1);
    gemm(einbuf, 32, W1aH, 64, b1aP, E, 64, 32, 64, 0, hA, nullptr, 64,
         nullptr, 1, nullptr, 0);
    gemm(hA, 64, W1bH, 128, b1bP, E, 128, 64, 128, 2, nullptr, nullptr, 0,
         valid1 + (size_t)g * E, EPQ, x1f + (size_t)g * M1 * 128, 128);
  }

  // ---- SA level 2 ----
  k_fps<<<BG, 512, 3 * M1 * 4, stream>>>(idx2, p1, M1, M2);
  k_gather_pos<<<blocks(BG * M2), 256, 0, stream>>>(p2, p1, idx2, M2, M1, BG);
  k_topk<<<blocks(BG * M2), 256, 0, stream>>>(cols2, valid2, p1, idx2, BG, M1, M2, 0.16f);
  k_fill_f16<<<blocks(EINB), 256, 0, stream>>>(einbuf, EINB);
  k_fill_f32<<<blocks((long long)BG * M2 * 256), 256, 0, stream>>>(x2f, (long long)BG * M2 * 256, NEGINF);
  for (int g = 0; g < BG; ++g) {
    int E = M2 * EPQ;   // 33792
    k_gather_sa<<<blocks((long long)E * 130), 256, 0, stream>>>(
        einbuf, 160, x1f + (size_t)g * M1 * 128, 128, p1 + (size_t)g * M1 * 2,
        idx2 + (size_t)g * M2, cols2 + (size_t)g * M2 * EPQ, M2);
    gemm(einbuf, 160, W2aH, 128, b2aP, E, 128, 160, 128, 0, hA, nullptr, 128,
         nullptr, 1, nullptr, 0);
    gemm(hA, 128, W2bH, 256, b2bP, E, 256, 128, 256, 2, nullptr, nullptr, 0,
         valid2 + (size_t)g * E, EPQ, x2f + (size_t)g * M2 * 256, 256);
  }

  // ---- mlp3 + global max pool (fused into epilogue, group = graph) ----
  k_fill_f16<<<blocks(EINB), 256, 0, stream>>>(einbuf, EINB);
  k_cvt_cols<<<blocks((long long)BG * M2 * 256), 256, 0, stream>>>(einbuf, 288, 0, x2f, 256, BG * M2, 256);
  k_cvt_cols<<<blocks((long long)BG * M2 * 2), 256, 0, stream>>>(einbuf, 288, 256, p2, 2, BG * M2, 2);
  gemm(einbuf, 288, W3aH, 256, b3aP, BG * M2, 256, 288, 256, 0, hA, nullptr, 256,
       nullptr, 1, nullptr, 0);
  k_fill_f32<<<blocks((long long)BG * 1024), 256, 0, stream>>>(x3f, (long long)BG * 1024, NEGINF);
  gemm(hA, 256, W3bH, 1024, b3bP, BG * M2, 1024, 256, 1024, 2, nullptr, nullptr, 0,
       nullptr, M2, x3f, 1024);

  // ---- fp3: u3 is x3 broadcast (single level-3 point) ----
  k_bcast_x3<<<blocks((long long)BG * M2 * 1024), 256, 0, stream>>>(einbuf, 1280, x3f, BG * M2, 1024, M2);
  k_cvt_cols<<<blocks((long long)BG * M2 * 256), 256, 0, stream>>>(einbuf, 1280, 1024, x2f, 256, BG * M2, 256);
  gemm(einbuf, 1280, Wf3H, 256, bf3P, BG * M2, 256, 1280, 256, 1, nullptr, d3f, 256,
       nullptr, 1, nullptr, 0);

  // ---- fp2: knn(8) p2 -> p1, concat x1 ----
  k_knn_search<<<blocks(BG * M1), 256, 0, stream>>>(kidx, kw, p1, p2, BG, M1, M2, 8);
  k_u_gather<<<blocks((long long)BG * M1 * 256), 256, 0, stream>>>(einbuf, 384, 0, d3f, 256, 256, kidx, kw, 8, BG * M1);
  k_cvt_cols<<<blocks((long long)BG * M1 * 128), 256, 0, stream>>>(einbuf, 384, 256, x1f, 128, BG * M1, 128);
  gemm(einbuf, 384, Wf2H, 128, bf2P, BG * M1, 128, 384, 128, 1, nullptr, d2f, 128,
       nullptr, 1, nullptr, 0);

  // ---- fp1: knn(16) p1 -> pb, concat x0, 3-layer MLP, store [B*N0,3] to d_out ----
  k_knn_search<<<blocks(BG * N0), 256, 0, stream>>>(kidx, kw, pos, p1, BG, N0, M1, 16);
  k_fill_f16<<<blocks(EINB), 256, 0, stream>>>(einbuf, EINB);
  k_u_gather<<<blocks((long long)BG * N0 * 128), 256, 0, stream>>>(einbuf, 160, 0, d2f, 128, 128, kidx, kw, 16, BG * N0);
  k_cvt_cols<<<blocks((long long)BG * N0 * 8), 256, 0, stream>>>(einbuf, 160, 128, x0f, 8, BG * N0, 8);
  gemm(einbuf, 160, Wf1aH, 128, bf1aP, BG * N0, 128, 160, 128, 0, hA, nullptr, 128,
       nullptr, 1, nullptr, 0);
  gemm(hA, 128, Wf1bH, 128, bf1bP, BG * N0, 128, 128, 128, 0, hB, nullptr, 128,
       nullptr, 1, nullptr, 0);
  gemm(hB, 128, Wf1cH, 16, bf1cP, BG * N0, 16, 128, 3, 1, nullptr, (float*)d_out, 3,
       nullptr, 1, nullptr, 0);
}
